// GridDecoder_78176994722689
// MI455X (gfx1250) — matrix operand order repaired and checked by reading, not compiled, not run on hardware
//
#include <hip/hip_runtime.h>
#include <hip/hip_bf16.h>
#include <math.h>

// ---------------- types / helpers (gfx1250 wave32) ----------------
typedef __attribute__((ext_vector_type(16))) __bf16    v16bf;
typedef __attribute__((ext_vector_type(8)))  __bf16    v8bf;
typedef __attribute__((ext_vector_type(8)))  float     v8f;
typedef __attribute__((ext_vector_type(8)))  unsigned  v8u;
typedef __attribute__((ext_vector_type(4)))  unsigned  v4u;

// Hardware packed f32->bf16 conversion (2 elements / VALU op).
__device__ __forceinline__ unsigned pk2(float lo, float hi) {
    unsigned d;
    asm("v_cvt_pk_bf16_f32 %0, %1, %2" : "=v"(d) : "v"(lo), "v"(hi));
    return d;
}
// Packed bf16 add / relu (2 elements per VALU op) — VOP3P V_PK_ADD_BF16 / V_PK_MAX_NUM_BF16.
__device__ __forceinline__ unsigned pk_add_bf16(unsigned a, unsigned b) {
    unsigned d;
    asm("v_pk_add_bf16 %0, %1, %2" : "=v"(d) : "v"(a), "v"(b));
    return d;
}
__device__ __forceinline__ unsigned pk_relu_bf16(unsigned a) {
    unsigned d, z = 0;
    asm("v_pk_max_num_bf16 %0, %1, %2" : "=v"(d) : "v"(a), "v"(z));
    return d;
}

// Scalar fallback (cold paths: GEMM result stores etc.)
__device__ __forceinline__ __bf16 f2bf(float f) {
    unsigned u = __builtin_bit_cast(unsigned, f);
    unsigned r = u + 0x7FFFu + ((u >> 16) & 1u);
    unsigned short s = (unsigned short)(r >> 16);
    return __builtin_bit_cast(__bf16, s);
}

// 16 contiguous bf16 (two aligned 16B loads) -> B fragment (element i <-> k = hf*16+i)
__device__ __forceinline__ v16bf load16(const __bf16* p) {
    v8bf lo = *(const v8bf*)p;
    v8bf hi = *(const v8bf*)(p + 8);
    return __builtin_shufflevector(lo, hi, 0,1,2,3,4,5,6,7,8,9,10,11,12,13,14,15);
}
// A fragment: runs [k0+hf*8, +8) and [k0+16+hf*8, +8)
__device__ __forceinline__ v16bf loadA(const __bf16* p) {
    v8bf lo = *(const v8bf*)p;
    v8bf hi = *(const v8bf*)(p + 16);
    return __builtin_shufflevector(lo, hi, 0,1,2,3,4,5,6,7,8,9,10,11,12,13,14,15);
}

// ---------------- elementwise f32 -> bf16 (packed) ----------------
__global__ void cvt_f32_to_bf16(const float* __restrict__ x, __bf16* __restrict__ y, int count)
{
    const int t = blockIdx.x * 256 + threadIdx.x;
    const int i = 2 * t;
    if (i < count) ((unsigned*)y)[t] = pk2(x[i], x[i + 1]);   // counts are even
}

// ---------------- pib = bf16(pi + b1), D=128 rows ----------------
__global__ void add_bias_cvt_bf16(const float* __restrict__ p, const float* __restrict__ b1,
                                  __bf16* __restrict__ o, int count)
{
    const int t = blockIdx.x * 256 + threadIdx.x;
    const int i = 2 * t;
    if (i < count)
        ((unsigned*)o)[t] = pk2(p[i] + b1[i & 127], p[i + 1] + b1[(i + 1) & 127]);
}

// ---------------- per-head V transpose: Vt[h][d][n] = V[h][n][d] ----------------
__global__ void transpose_v(const __bf16* __restrict__ qkvb, __bf16* __restrict__ vt)
{
    const int t = blockIdx.x * 256 + threadIdx.x;     // 4*32*1024 total
    const int n = t & 1023;
    const int d = (t >> 10) & 31;
    const int h = t >> 15;
    vt[(size_t)h * 32 * 1024 + (size_t)d * 1024 + n] =
        qkvb[(size_t)n * 384 + 256 + h * 32 + d];
}

// ---------------- WMMA GEMM: bf16 inputs, f32 accumulate ----------------
// Y[m,n] = act( scale * sum_k X[m,k]*W[n,k] + bias[n] ), W always (N x K) row-major.
// One wave computes a 16 x (16*NT) tile. Batched over blockIdx.z (element strides).
template<int NT, bool OUTBF>
__global__ void gemm_bf16(const __bf16* __restrict__ X, int ldx, long long sxb,
                          const __bf16* __restrict__ W, int ldw, long long swb,
                          const float* __restrict__ bias,
                          void* __restrict__ Yv, int ldy, long long syb,
                          int K, int act, float scale)
{
    const int z = blockIdx.z;
    X += (size_t)z * sxb;  W += (size_t)z * swb;

    const int n0   = blockIdx.x * 16 * NT;
    const int m0   = blockIdx.y * 16;
    const int lane = threadIdx.x & 31;
    const int l15  = lane & 15;
    const int hf   = lane >> 4;

    v8f c[NT];
    #pragma unroll
    for (int t = 0; t < NT; ++t) c[t] = {};

    for (int k0 = 0; k0 < K; k0 += 32) {
        const v16bf a = load16(X + (size_t)(m0 + l15) * ldx + k0 + hf * 16);
        #pragma unroll
        for (int t = 0; t < NT; ++t) {
            const v16bf b = load16(W + (size_t)(n0 + t * 16 + l15) * ldw + k0 + hf * 16);
            c[t] = __builtin_amdgcn_wmma_f32_16x16x32_bf16(false, a, false, b,
                                                           (short)0, c[t], false, false);
        }
    }

    #pragma unroll
    for (int t = 0; t < NT; ++t) {
        const int   n  = n0 + t * 16 + l15;
        const float bn = bias ? bias[n] : 0.0f;
        #pragma unroll
        for (int r = 0; r < 8; ++r) {
            float v = c[t][r] * scale + bn;
            if (act == 1) v = fmaxf(v, 0.0f);
            const size_t yi = (size_t)z * syb + (size_t)(m0 + 8 * hf + r) * ldy + n;
            if (OUTBF) ((__bf16*)Yv)[yi] = f2bf(v);
            else       ((float*)Yv)[yi]  = v;
        }
    }
}

// ---------------- row softmax: f32 in, bf16 out (1024 cols, 256 thr) ----------------
__global__ void softmax_rows_1024(const float* __restrict__ S, __bf16* __restrict__ Sb)
{
    const float* r = S  + (size_t)blockIdx.x * 1024;
    __bf16*      o = Sb + (size_t)blockIdx.x * 1024;
    const int t = threadIdx.x;

    float v[4];
    float mx = -INFINITY;
    #pragma unroll
    for (int j = 0; j < 4; ++j) { v[j] = r[t + j * 256]; mx = fmaxf(mx, v[j]); }
    #pragma unroll
    for (int m = 16; m >= 1; m >>= 1) mx = fmaxf(mx, __shfl_xor(mx, m, 32));
    __shared__ float red[8];
    if ((t & 31) == 0) red[t >> 5] = mx;
    __syncthreads();
    mx = red[0];
    #pragma unroll
    for (int w = 1; w < 8; ++w) mx = fmaxf(mx, red[w]);
    __syncthreads();

    float sum = 0.0f;
    #pragma unroll
    for (int j = 0; j < 4; ++j) { v[j] = __expf(v[j] - mx); sum += v[j]; }
    #pragma unroll
    for (int m = 16; m >= 1; m >>= 1) sum += __shfl_xor(sum, m, 32);
    if ((t & 31) == 0) red[t >> 5] = sum;
    __syncthreads();
    sum = 0.0f;
    #pragma unroll
    for (int w = 0; w < 8; ++w) sum += red[w];

    const float inv = 1.0f / sum;
    #pragma unroll
    for (int j = 0; j < 4; ++j) o[t + j * 256] = f2bf(v[j] * inv);
}

// ---------------- fused residual + LayerNorm, writes f32 x and bf16 xb ----------------
__global__ void add_layernorm_128(float* __restrict__ x, __bf16* __restrict__ xb,
                                  const float* __restrict__ res,
                                  const float* __restrict__ w, const float* __restrict__ b)
{
    const int t = threadIdx.x;               // 128 threads = 4 waves
    const size_t idx = (size_t)blockIdx.x * 128 + t;

    float v = x[idx] + res[idx];

    __shared__ float sm[4], sv[4];
    float s = v;
    #pragma unroll
    for (int m = 16; m >= 1; m >>= 1) s += __shfl_xor(s, m, 32);
    if ((t & 31) == 0) sm[t >> 5] = s;
    __syncthreads();
    const float mean = (sm[0] + sm[1] + sm[2] + sm[3]) * (1.0f / 128.0f);

    const float d = v - mean;
    float q = d * d;
    #pragma unroll
    for (int m = 16; m >= 1; m >>= 1) q += __shfl_xor(q, m, 32);
    if ((t & 31) == 0) sv[t >> 5] = q;
    __syncthreads();
    const float var = (sv[0] + sv[1] + sv[2] + sv[3]) * (1.0f / 128.0f);

    const float out = d * rsqrtf(var + 1e-5f) * w[t] + b[t];
    x[idx]  = out;
    xb[idx] = f2bf(out);
}

// ---------------- fused pairwise decoder (all-bf16 fragment build) ----------------
// logits[i,j,c] = sum_h relu(pib[i,h] + pjb[j,h]) * w2[c,h] + b2[c]   (b1 folded into pib)
// One wave per (i, 64-wide j block): w2 B frags preloaded; pi dwords hoisted per k-step;
// A fragments built with packed v_pk_add_bf16 / v_pk_max_num_bf16 (2 elems per op).
__global__ void pair_decoder_wmma(const __bf16* __restrict__ pib, const __bf16* __restrict__ pjb,
                                  const __bf16* __restrict__ w2, const float* __restrict__ b2,
                                  float* __restrict__ out)
{
    const int i    = blockIdx.y;
    const int j0   = blockIdx.x * 64;
    const int lane = threadIdx.x & 31;
    const int l15  = lane & 15;
    const int hf   = lane >> 4;

    // B fragments: w2 padded to 16 cols (lanes with c>=5 contribute zeros)
    v16bf bw[4];
    #pragma unroll
    for (int ks4 = 0; ks4 < 4; ++ks4) {
        if (l15 < 5) bw[ks4] = load16(w2 + (size_t)l15 * 128 + ks4 * 32 + hf * 16);
        else         bw[ks4] = __builtin_bit_cast(v16bf, (v8u)0);
    }

    v8f acc[4];
    #pragma unroll
    for (int jt = 0; jt < 4; ++jt) acc[jt] = {};

    const __bf16* piB = pib + (size_t)i * 128;
    #pragma unroll
    for (int ks4 = 0; ks4 < 4; ++ks4) {
        const int ks = ks4 * 32;
        const v4u pl = *(const v4u*)(piB + ks + hf * 16);       // k = ks+hf*16   .. +8
        const v4u ph = *(const v4u*)(piB + ks + hf * 16 + 8);   // k = ks+hf*16+8 .. +8
        #pragma unroll
        for (int jt = 0; jt < 4; ++jt) {
            const __bf16* pjr = pjb + (size_t)(j0 + jt * 16 + l15) * 128 + ks + hf * 16;
            const v4u jl = *(const v4u*)(pjr);
            const v4u jh = *(const v4u*)(pjr + 8);
            v8u au;
            #pragma unroll
            for (int e2 = 0; e2 < 4; ++e2) au[e2]     = pk_relu_bf16(pk_add_bf16(pl[e2], jl[e2]));
            #pragma unroll
            for (int e2 = 0; e2 < 4; ++e2) au[4 + e2] = pk_relu_bf16(pk_add_bf16(ph[e2], jh[e2]));
            const v16bf a = __builtin_bit_cast(v16bf, au);
            acc[jt] = __builtin_amdgcn_wmma_f32_16x16x32_bf16(false, a, false, bw[ks4],
                                                              (short)0, acc[jt], false, false);
        }
    }

    if (l15 < 5) {
        const float bc = b2[l15];
        #pragma unroll
        for (int jt = 0; jt < 4; ++jt) {
            #pragma unroll
            for (int r = 0; r < 8; ++r) {
                const size_t j = (size_t)j0 + jt * 16 + 8 * hf + r;
                out[((size_t)i * 1024 + j) * 5 + l15] = acc[jt][r] + bc;
            }
        }
    }
}

// ---------------- host orchestration ----------------
extern "C" void kernel_launch(void* const* d_in, const int* in_sizes, int n_in,
                              void* d_out, int out_size, void* d_ws, size_t ws_size,
                              hipStream_t stream)
{
    (void)in_sizes; (void)n_in; (void)out_size; (void)ws_size;
    const int N = 1024, D = 128, H = 4, DH = 32, L = 4, FF = 2048;

    const float* node    = (const float*)d_in[0];
    const float* qkv_w   = (const float*)d_in[1];
    const float* qkv_b   = (const float*)d_in[2];
    const float* out_w   = (const float*)d_in[3];
    const float* out_b   = (const float*)d_in[4];
    const float* ln1_w   = (const float*)d_in[5];
    const float* ln1_b   = (const float*)d_in[6];
    const float* ff1_w   = (const float*)d_in[7];
    const float* ff1_b   = (const float*)d_in[8];
    const float* ff2_w   = (const float*)d_in[9];
    const float* ff2_b   = (const float*)d_in[10];
    const float* ln2_w   = (const float*)d_in[11];
    const float* ln2_b   = (const float*)d_in[12];
    const float* pair_w1 = (const float*)d_in[13];
    const float* pair_b1 = (const float*)d_in[14];
    const float* pair_w2 = (const float*)d_in[15];
    const float* pair_b2 = (const float*)d_in[16];

    // ---- workspace layout (bump allocator) ----
    char* wp = (char*)d_ws;
    auto allocF = [&](size_t n) { float*  p = (float*)wp;  wp += n * 4; return p; };
    auto allocB = [&](size_t n) { __bf16* p = (__bf16*)wp; wp += n * 2; return p; };

    float*  x     = allocF((size_t)N * D);
    float*  S     = allocF((size_t)H * N * N);
    float*  tmp   = allocF((size_t)N * D);
    float*  pi    = allocF((size_t)N * D);
    float*  pj    = allocF((size_t)N * D);
    __bf16* xb    = allocB((size_t)N * D);
    __bf16* qkvb  = allocB((size_t)N * 3 * D);
    __bf16* Sb    = allocB((size_t)H * N * N);
    __bf16* Vt    = allocB((size_t)H * DH * N);
    __bf16* ctxb  = allocB((size_t)N * D);
    __bf16* ffb   = allocB((size_t)N * FF);
    __bf16* pib   = allocB((size_t)N * D);
    __bf16* pjb   = allocB((size_t)N * D);
    __bf16* wqkvb = allocB((size_t)L * 3 * D * D);
    __bf16* woutb = allocB((size_t)L * D * D);
    __bf16* wff1b = allocB((size_t)L * FF * D);
    __bf16* wff2b = allocB((size_t)L * D * FF);
    __bf16* wp1b  = allocB((size_t)D * 2 * D);
    __bf16* wp2b  = allocB((size_t)5 * D);

    auto cvt = [&](const float* s, __bf16* d, int cnt) {
        cvt_f32_to_bf16<<<dim3((unsigned)((cnt / 2 + 255) / 256)), dim3(256), 0, stream>>>(s, d, cnt);
    };

    hipMemcpyAsync(x, node, (size_t)N * D * sizeof(float), hipMemcpyDeviceToDevice, stream);
    cvt(node, xb, N * D);
    cvt(qkv_w,   wqkvb, L * 3 * D * D);
    cvt(out_w,   woutb, L * D * D);
    cvt(ff1_w,   wff1b, L * FF * D);
    cvt(ff2_w,   wff2b, L * D * FF);
    cvt(pair_w1, wp1b,  D * 2 * D);
    cvt(pair_w2, wp2b,  5 * D);

    const dim3 wave(32);
    const float inv_sqrt_dh = 0.17677669529663689f; // 1/sqrt(32)

    for (int l = 0; l < L; ++l) {
        // QKV -> bf16 (N x 3D)
        gemm_bf16<4, true><<<dim3(3 * D / 64, N / 16, 1), wave, 0, stream>>>(
            xb, D, 0, wqkvb + (size_t)l * 3 * D * D, D, 0,
            qkv_b + (size_t)l * 3 * D, qkvb, 3 * D, 0, D, 0, 1.0f);

        // scores[h] = Q_h @ K_h^T / sqrt(DH) -> f32
        gemm_bf16<4, false><<<dim3(N / 64, N / 16, H), wave, 0, stream>>>(
            qkvb, 3 * D, DH, qkvb + D, 3 * D, DH,
            nullptr, S, N, (long long)N * N, DH, 0, inv_sqrt_dh);

        softmax_rows_1024<<<dim3(H * N), dim3(256), 0, stream>>>(S, Sb);

        // per-head V transpose -> Vt (H x DH x N)
        transpose_v<<<dim3(H * DH * N / 256), dim3(256), 0, stream>>>(qkvb, Vt);

        // ctx[h] = P_h @ V_h -> bf16 (normal-mode B via Vt)
        gemm_bf16<2, true><<<dim3(1, N / 16, H), wave, 0, stream>>>(
            Sb, N, (long long)N * N, Vt, N, (long long)DH * N,
            nullptr, ctxb, D, DH, N, 0, 1.0f);

        // attn_out = ctx @ out_w^T + out_b -> f32
        gemm_bf16<4, false><<<dim3(D / 64, N / 16, 1), wave, 0, stream>>>(
            ctxb, D, 0, woutb + (size_t)l * D * D, D, 0,
            out_b + (size_t)l * D, tmp, D, 0, D, 0, 1.0f);

        add_layernorm_128<<<dim3(N), dim3(128), 0, stream>>>(
            x, xb, tmp, ln1_w + (size_t)l * D, ln1_b + (size_t)l * D);

        // FF1 (relu) -> bf16
        gemm_bf16<4, true><<<dim3(FF / 64, N / 16, 1), wave, 0, stream>>>(
            xb, D, 0, wff1b + (size_t)l * FF * D, D, 0,
            ff1_b + (size_t)l * FF, ffb, FF, 0, D, 1, 1.0f);

        // FF2 -> f32
        gemm_bf16<4, false><<<dim3(D / 64, N / 16, 1), wave, 0, stream>>>(
            ffb, FF, 0, wff2b + (size_t)l * D * FF, FF, 0,
            ff2_b + (size_t)l * D, tmp, D, 0, FF, 0, 1.0f);

        add_layernorm_128<<<dim3(N), dim3(128), 0, stream>>>(
            x, xb, tmp, ln2_w + (size_t)l * D, ln2_b + (size_t)l * D);
    }

    // pi = t @ wi^T, pj = t @ wj^T (column halves of pair_w1, ldw = 2D) -> f32
    gemm_bf16<4, false><<<dim3(D / 64, N / 16, 1), wave, 0, stream>>>(
        xb, D, 0, wp1b, 2 * D, 0, nullptr, pi, D, 0, D, 0, 1.0f);
    gemm_bf16<4, false><<<dim3(D / 64, N / 16, 1), wave, 0, stream>>>(
        xb, D, 0, wp1b + D, 2 * D, 0, nullptr, pj, D, 0, D, 0, 1.0f);

    // pib = bf16(pi + b1); pjb = bf16(pj)
    add_bias_cvt_bf16<<<dim3((N * D / 2 + 255) / 256), dim3(256), 0, stream>>>(
        pi, pair_b1, pib, N * D);
    cvt(pj, pjb, N * D);

    // fused pairwise decoder -> logits (N,N,5)
    pair_decoder_wmma<<<dim3(N / 64, N), wave, 0, stream>>>(
        pib, pjb, wp2b, pair_b2, (float*)d_out);
}
